// ExtractNetwork_37752762531923
// MI455X (gfx1250) — compile-verified, matrix operand order checked by reading, not run on hardware
//
#include <hip/hip_runtime.h>
#include <hip/hip_bf16.h>
#include <cstddef>
#include <cstdint>

#define N_NODES 100000
#define DEG 16
#define HID 256
#define EPS_BN 1e-5f
#define COS_S 4.6f

typedef __attribute__((ext_vector_type(16))) __bf16 v16bf;
typedef __attribute__((ext_vector_type(8)))  float  v8f;

union FragBF {
    v16bf v;
    uint4 q[2];
};

// ---------------------------------------------------------------------------
// Workspace layout (byte offsets). hc (100000x768 f32) overlaps y1 (dead after conv2).
// ---------------------------------------------------------------------------
#define WS_STATS   ((size_t)0)                       // 3584 floats (bn1/bn2/dbn sums+scale/shift)
#define WS_HC      ((size_t)16384)                   // 100000*768*4 = 307,200,000 (also y1: 100000*384*4)
#define WS_INBF    (WS_HC + (size_t)307200000)       // 100000*256*2 bf16
#define WS_FEAT    (WS_INBF + (size_t)51200000)      // 100000*256*4 f32 (also raw y2)
#define WS_ELER    (WS_FEAT + (size_t)102400000)     // 100000*4 f32
#define WS_HCNBF   (WS_ELER + (size_t)1600000)       // 100000*768*2 bf16
#define WS_ROWNORM (WS_HCNBF + (size_t)153600000)    // 100000 f32
#define WS_FC1BF   (WS_ROWNORM + (size_t)400000)     // 65536 bf16
#define WS_FC2BF   (WS_FC1BF + (size_t)131072)       // 65536 bf16
#define WS_COSWN   (WS_FC2BF + (size_t)131072)       // 256*768 bf16

// stats float offsets
#define ST_BN1  0      // sum[32], sq[32], scale[32], shift[32]
#define ST_BN2  128    // sum[64], sq[64], scale[64], shift[64]
#define ST_DBN  512    // sum[768], sq[768], scale[768], shift[768]

// ---------------------------------------------------------------------------
// Small helpers
// ---------------------------------------------------------------------------
__global__ void f2bf_kernel(const float* __restrict__ in, __bf16* __restrict__ out, int n) {
    int i = blockIdx.x * blockDim.x + threadIdx.x;
    if (i < n) out[i] = (__bf16)in[i];
}

// L2-normalize each 768-wide row of cos_W (256 rows) -> bf16
__global__ void cosw_norm_kernel(const float* __restrict__ W, __bf16* __restrict__ Wn) {
    int r = blockIdx.x;
    int tid = threadIdx.x;
    __shared__ float red[8];
    __shared__ float s_inv;
    float v[3];
    float ss = 0.f;
#pragma unroll
    for (int t = 0; t < 3; t++) {
        int c = tid + t * 256;
        v[t] = W[(size_t)r * 768 + c];
        ss += v[t] * v[t];
    }
    for (int off = 16; off >= 1; off >>= 1) ss += __shfl_xor(ss, off, 32);
    if ((tid & 31) == 0) red[tid >> 5] = ss;
    __syncthreads();
    if (tid == 0) {
        float tot = 0.f;
#pragma unroll
        for (int k = 0; k < 8; k++) tot += red[k];
        s_inv = 1.f / fmaxf(sqrtf(tot), 1e-12f);
    }
    __syncthreads();
#pragma unroll
    for (int t = 0; t < 3; t++) {
        int c = tid + t * 256;
        Wn[(size_t)r * 768 + c] = (__bf16)(v[t] * s_inv);
    }
}

// ---------------------------------------------------------------------------
// Conv1: (N,60) -> (N,32,12), kernel 5 stride 5 (non-overlapping) + bias.
// Accumulate per-channel sum/sumsq for BN1.
// ---------------------------------------------------------------------------
__global__ void conv1_kernel(const float* __restrict__ x, const float* __restrict__ w1,
                             const float* __restrict__ b1, float* __restrict__ y1,
                             float* __restrict__ stats) {
    __shared__ float sw[160];
    __shared__ float sb[32];
    __shared__ float ssum[32], ssq[32];
    int tid = threadIdx.x;
    if (tid < 160) sw[tid] = w1[tid];
    if (tid < 32) { sb[tid] = b1[tid]; ssum[tid] = 0.f; ssq[tid] = 0.f; }
    __syncthreads();

    float psum[32], psq[32];
#pragma unroll
    for (int c = 0; c < 32; c++) { psum[c] = 0.f; psq[c] = 0.f; }

    int n = blockIdx.x * blockDim.x + tid;
    if (n < N_NODES) {
        const float* xr = x + (size_t)n * 60;
        float* yr = y1 + (size_t)n * 384;
        for (int p = 0; p < 12; p++) {
            float xv[5];
#pragma unroll
            for (int k = 0; k < 5; k++) xv[k] = xr[p * 5 + k];
#pragma unroll
            for (int c = 0; c < 32; c++) {
                float acc = sb[c];
#pragma unroll
                for (int k = 0; k < 5; k++) acc += sw[c * 5 + k] * xv[k];
                yr[c * 12 + p] = acc;
                psum[c] += acc;
                psq[c] += acc * acc;
            }
        }
    }
#pragma unroll
    for (int c = 0; c < 32; c++) {
        atomicAdd(&ssum[c], psum[c]);
        atomicAdd(&ssq[c], psq[c]);
    }
    __syncthreads();
    if (tid < 32) {
        atomicAdd(&stats[ST_BN1 + tid], ssum[tid]);
        atomicAdd(&stats[ST_BN1 + 32 + tid], ssq[tid]);
    }
}

// Finalize BN: base[0..C)=sum, [C..2C)=sq -> write [2C..3C)=scale, [3C..4C)=shift
__global__ void finalize_bn_kernel(float* __restrict__ base, const float* __restrict__ g,
                                   const float* __restrict__ b, int C, float count) {
    int c = blockIdx.x * blockDim.x + threadIdx.x;
    if (c < C) {
        float mean = base[c] / count;
        float var = base[C + c] / count - mean * mean;
        float inv = rsqrtf(var + EPS_BN);
        float sc = g[c] * inv;
        base[2 * C + c] = sc;
        base[3 * C + c] = b[c] - mean * sc;
    }
}

// ---------------------------------------------------------------------------
// Conv2: BN1+ReLU on y1, then (32,12) -> (64,4) kernel 3 stride 3.
// Block = 4 nodes x 64 out channels. Accumulate BN2 stats.
// y2 stored as h layout: [n][oc*4+p]  (== reshape(N,256))
// ---------------------------------------------------------------------------
__global__ void conv2_kernel(const float* __restrict__ y1, const float* __restrict__ w2,
                             const float* __restrict__ b2, const float* __restrict__ stats,
                             float* __restrict__ y2, float* __restrict__ statsg) {
    __shared__ float sw[6144];            // w2 [oc][ic][k]
    __shared__ float tile[4][12][33];     // [node][pos][ic]
    __shared__ float sb2[64];
    __shared__ float s_sc[32], s_sh[32];
    __shared__ float s2sum[64], s2sq[64];

    int tid = threadIdx.x;
    int n0 = blockIdx.x * 4;

    for (int idx = tid; idx < 6144; idx += 256) sw[idx] = w2[idx];
    if (tid < 64) { sb2[tid] = b2[tid]; s2sum[tid] = 0.f; s2sq[tid] = 0.f; }
    if (tid < 32) { s_sc[tid] = stats[ST_BN1 + 64 + tid]; s_sh[tid] = stats[ST_BN1 + 96 + tid]; }
    __syncthreads();

    for (int idx = tid; idx < 4 * 384; idx += 256) {
        int nd = idx / 384;
        int r = idx - nd * 384;
        int ic = r / 12;
        int p = r - ic * 12;
        float v = y1[(size_t)(n0 + nd) * 384 + r];
        v = fmaxf(v * s_sc[ic] + s_sh[ic], 0.f);
        tile[nd][p][ic] = v;
    }
    __syncthreads();

    int nd = tid >> 6;
    int oc = tid & 63;
    float ssum = 0.f, ssq = 0.f;
    float* outp = y2 + (size_t)(n0 + nd) * 256 + oc * 4;
#pragma unroll
    for (int p = 0; p < 4; p++) {
        float acc = sb2[oc];
        const float* wp = &sw[oc * 96];
#pragma unroll
        for (int ic = 0; ic < 32; ic++) {
            acc += wp[ic * 3 + 0] * tile[nd][3 * p + 0][ic];
            acc += wp[ic * 3 + 1] * tile[nd][3 * p + 1][ic];
            acc += wp[ic * 3 + 2] * tile[nd][3 * p + 2][ic];
        }
        outp[p] = acc;
        ssum += acc;
        ssq += acc * acc;
    }
    atomicAdd(&s2sum[oc], ssum);
    atomicAdd(&s2sq[oc], ssq);
    __syncthreads();
    if (tid < 64) {
        atomicAdd(&statsg[ST_BN2 + tid], s2sum[tid]);
        atomicAdd(&statsg[ST_BN2 + 64 + tid], s2sq[tid]);
    }
}

// Apply BN2 + ReLU to raw y2 -> bf16 activations
__global__ void bn2relu_bf_kernel(const float* __restrict__ y2, const float* __restrict__ stats,
                                  __bf16* __restrict__ out) {
    size_t i = (size_t)blockIdx.x * blockDim.x + threadIdx.x;
    if (i < (size_t)N_NODES * 256) {
        int c = ((int)(i & 255)) >> 2;   // channel = (idx%256)/4
        float v = y2[i] * stats[ST_BN2 + 128 + c] + stats[ST_BN2 + 192 + c];
        out[i] = (__bf16)fmaxf(v, 0.f);
    }
}

// ---------------------------------------------------------------------------
// WMMA bf16 GEMM:  C[M x NC] = A[M x K] * B[NC x K]^T   (B row-major, row = out col)
// Block = 256 threads = 8 waves (2 M-tiles x 4 wave-columns).
// Each wave owns a 16x64 output strip: one A fragment reused across 4 B
// fragments -> 4 independent accumulators per wave (raises FLOP/byte from
// ~8 to ~13 against the L2-resident operands; no RAW between WMMAs).
// NC must be a multiple of 256; block covers full N (grid.x = NC/256).
// Optional per-row scale: mult / max(rowscale[row],1e-12).
// ---------------------------------------------------------------------------
__global__ void wmma_gemm_kernel(const __bf16* __restrict__ A, const __bf16* __restrict__ B,
                                 float* __restrict__ C, int K, int NC,
                                 const float* __restrict__ rowscale, float mult) {
    int tid = threadIdx.x;
    int w = tid >> 5;
    int lane = tid & 31;
    int lidx = lane & 15;
    int lh = lane >> 4;

    int m0 = blockIdx.y * 32 + (w >> 2) * 16;
    int n0 = blockIdx.x * 256 + (w & 3) * 64;

    const __bf16* Arow = A + (size_t)(m0 + lidx) * K;
    const __bf16* Brow = B + (size_t)(n0 + lidx) * K;

    v8f acc0 = {}, acc1 = {}, acc2 = {}, acc3 = {};
    for (int k0 = 0; k0 < K; k0 += 32) {
        // A fragment (16x32 bf16): lanes 0-15 row M=lidx hold K {0..7,16..23},
        // lanes 16-31 hold K {8..15,24..31}  -> two b128 loads
        FragBF a;
        a.q[0] = *(const uint4*)(Arow + k0 + lh * 8);
        a.q[1] = *(const uint4*)(Arow + k0 + 16 + lh * 8);
        // 4 B fragments (32x16 each), mirrored layout with N as lane dim
        FragBF b0, b1, b2, b3;
        const __bf16* bp = Brow + k0 + lh * 8;
        b0.q[0] = *(const uint4*)(bp);
        b0.q[1] = *(const uint4*)(bp + 16);
        b1.q[0] = *(const uint4*)(bp + (size_t)16 * K);
        b1.q[1] = *(const uint4*)(bp + (size_t)16 * K + 16);
        b2.q[0] = *(const uint4*)(bp + (size_t)32 * K);
        b2.q[1] = *(const uint4*)(bp + (size_t)32 * K + 16);
        b3.q[0] = *(const uint4*)(bp + (size_t)48 * K);
        b3.q[1] = *(const uint4*)(bp + (size_t)48 * K + 16);
        acc0 = __builtin_amdgcn_wmma_f32_16x16x32_bf16(false, a.v, false, b0.v,
                                                       (short)0, acc0, false, false);
        acc1 = __builtin_amdgcn_wmma_f32_16x16x32_bf16(false, a.v, false, b1.v,
                                                       (short)0, acc1, false, false);
        acc2 = __builtin_amdgcn_wmma_f32_16x16x32_bf16(false, a.v, false, b2.v,
                                                       (short)0, acc2, false, false);
        acc3 = __builtin_amdgcn_wmma_f32_16x16x32_bf16(false, a.v, false, b3.v,
                                                       (short)0, acc3, false, false);
    }

#pragma unroll
    for (int r = 0; r < 8; r++) {
        int row = m0 + r + lh * 8;
        float s = rowscale ? (mult / fmaxf(rowscale[row], 1e-12f)) : mult;
        float* crow = C + (size_t)row * NC + n0 + lidx;
        crow[0]  = acc0[r] * s;
        crow[16] = acc1[r] * s;
        crow[32] = acc2[r] * s;
        crow[48] = acc3[r] * s;
    }
}

// ---------------------------------------------------------------------------
// el/er per node: dots of feat row halves with attn_l/attn_r (each (2,128))
// eler layout: [n][4] = el0, el1, er0, er1
// ---------------------------------------------------------------------------
__global__ void eler_kernel(const float* __restrict__ feat, const float* __restrict__ al,
                            const float* __restrict__ ar, float* __restrict__ eler) {
    __shared__ float s_al[256], s_ar[256];
    int tid = threadIdx.x;
    s_al[tid] = al[tid];
    s_ar[tid] = ar[tid];
    __syncthreads();
    int n = blockIdx.x * blockDim.x + tid;
    if (n >= N_NODES) return;
    const float* f = feat + (size_t)n * 256;
    float e0 = 0.f, e1 = 0.f, r0 = 0.f, r1 = 0.f;
    for (int d = 0; d < 128; d++) {
        float f0 = f[d], f1 = f[128 + d];
        e0 += f0 * s_al[d];       e1 += f1 * s_al[128 + d];
        r0 += f0 * s_ar[d];       r1 += f1 * s_ar[128 + d];
    }
    float4 v = make_float4(e0, e1, r0, r1);
    *(float4*)(eler + (size_t)n * 4) = v;
}

__device__ __forceinline__ float act_mode1(float x) {
    float r = fmaxf(x, 0.f);
    float t = tanhf(x);
    float ls = (x >= 0.f) ? -log1pf(__expf(-x)) : (x - log1pf(__expf(x)));
    return r + t + ls;
}

// ---------------------------------------------------------------------------
// GAT attention + aggregation. One block (128 thr) per destination node.
// Edges for node i are [16i,16i+16) (dst = repeat(arange(N),16)).
// Wave 0: lane = edge(0..15) x head(0..1): leaky-relu, 16-way softmax via shfl.
// Aggregation: thread t -> dims t (head0) and 128+t (head1).
// hc_out points at hc + layer column offset (row stride 768).
// ---------------------------------------------------------------------------
__global__ void gat_aggregate_kernel(const float* __restrict__ feat, const float* __restrict__ eler,
                                     const int* __restrict__ src, const float* __restrict__ bias,
                                     float* __restrict__ hc_out, __bf16* __restrict__ bfout,
                                     int mode) {
    int i = blockIdx.x;
    int tid = threadIdx.x;
    __shared__ float s_alpha[32];
    __shared__ int s_src[16];

    if (tid < 32) {
        int j = tid & 15;
        int hh = tid >> 4;
        int sj = src[i * DEG + j];
        if (hh == 0) s_src[j] = sj;
        float e = eler[(size_t)sj * 4 + hh] + eler[(size_t)i * 4 + 2 + hh];
        e = (e > 0.f) ? e : 0.2f * e;                 // leaky_relu 0.2
        float m = e;
        for (int off = 8; off >= 1; off >>= 1) m = fmaxf(m, __shfl_xor(m, off, 32));
        float ex = __expf(e - m);
        float den = ex;
        for (int off = 8; off >= 1; off >>= 1) den += __shfl_xor(den, off, 32);
        s_alpha[tid] = ex / fmaxf(den, 1e-12f);
    }
    __syncthreads();

    float acc0 = 0.f, acc1 = 0.f;
#pragma unroll 4
    for (int j = 0; j < 16; j++) {
        const float* fr = feat + (size_t)s_src[j] * 256;
        acc0 += s_alpha[j]      * fr[tid];
        acc1 += s_alpha[16 + j] * fr[128 + tid];
    }
    float g0 = acc0 + bias[tid];
    float g1 = acc1 + bias[128 + tid];
    float o0, o1;
    if (mode == 1)      { o0 = act_mode1(g0); o1 = act_mode1(g1); }
    else if (mode == 2) { o0 = tanhf(g0);     o1 = tanhf(g1); }
    else                { o0 = fmaxf(g0, 0.f); o1 = fmaxf(g1, 0.f); }
    hc_out[(size_t)i * 768 + tid] = o0;
    hc_out[(size_t)i * 768 + 128 + tid] = o1;
    if (bfout) {
        bfout[(size_t)i * 256 + tid] = (__bf16)o0;
        bfout[(size_t)i * 256 + 128 + tid] = (__bf16)o1;
    }
}

// Column stats for dbn: grid(3, rowchunks), thread = one column over 256 rows.
__global__ void colstats_kernel(const float* __restrict__ hc, float* __restrict__ stats) {
    int col = blockIdx.x * 256 + threadIdx.x;
    int r0 = blockIdx.y * 256;
    int r1 = min(r0 + 256, N_NODES);
    float s = 0.f, q = 0.f;
    for (int r = r0; r < r1; r++) {
        float v = hc[(size_t)r * 768 + col];
        s += v;
        q += v * v;
    }
    atomicAdd(&stats[ST_DBN + col], s);
    atomicAdd(&stats[ST_DBN + 768 + col], q);
}

// Apply dbn scale/shift, write bf16 hcn, compute row L2 norm.
__global__ void hcn_rownorm_kernel(const float* __restrict__ hc, const float* __restrict__ stats,
                                   __bf16* __restrict__ hcn, float* __restrict__ rn) {
    int i = blockIdx.x;
    int tid = threadIdx.x;
    __shared__ float red[8];
    float ss = 0.f;
#pragma unroll
    for (int t = 0; t < 3; t++) {
        int c = tid + t * 256;
        float v = hc[(size_t)i * 768 + c] * stats[ST_DBN + 1536 + c] + stats[ST_DBN + 2304 + c];
        hcn[(size_t)i * 768 + c] = (__bf16)v;
        ss += v * v;
    }
    for (int off = 16; off >= 1; off >>= 1) ss += __shfl_xor(ss, off, 32);
    if ((tid & 31) == 0) red[tid >> 5] = ss;
    __syncthreads();
    if (tid == 0) {
        float tot = 0.f;
#pragma unroll
        for (int k = 0; k < 8; k++) tot += red[k];
        rn[i] = sqrtf(tot);
    }
}

// ---------------------------------------------------------------------------
extern "C" void kernel_launch(void* const* d_in, const int* in_sizes, int n_in,
                              void* d_out, int out_size, void* d_ws, size_t ws_size,
                              hipStream_t stream) {
    (void)in_sizes; (void)n_in; (void)out_size; (void)ws_size;

    const float* x      = (const float*)d_in[0];
    const int*   src    = (const int*)d_in[1];
    // d_in[2] = dst: implicit (repeat(arange(N),16)), not needed
    const float* w1     = (const float*)d_in[3];
    const float* b1     = (const float*)d_in[4];
    const float* bn1_g  = (const float*)d_in[5];
    const float* bn1_b  = (const float*)d_in[6];
    const float* w2     = (const float*)d_in[7];
    const float* b2     = (const float*)d_in[8];
    const float* bn2_g  = (const float*)d_in[9];
    const float* bn2_b  = (const float*)d_in[10];
    const float* fc1_w  = (const float*)d_in[11];
    const float* attn1l = (const float*)d_in[12];
    const float* attn1r = (const float*)d_in[13];
    const float* bias1  = (const float*)d_in[14];
    const float* fc2_w  = (const float*)d_in[15];
    const float* attn2l = (const float*)d_in[16];
    const float* attn2r = (const float*)d_in[17];
    const float* bias2  = (const float*)d_in[18];
    const float* dbn_g  = (const float*)d_in[19];
    const float* dbn_b  = (const float*)d_in[20];
    const float* cos_W  = (const float*)d_in[21];

    char* ws = (char*)d_ws;
    float*  stats   = (float*)(ws + WS_STATS);
    float*  hc      = (float*)(ws + WS_HC);        // also y1 during conv stage
    float*  y1      = hc;
    __bf16* inbf    = (__bf16*)(ws + WS_INBF);
    float*  featbuf = (float*)(ws + WS_FEAT);      // also raw y2 during conv stage
    float*  elerbuf = (float*)(ws + WS_ELER);
    __bf16* hcnbf   = (__bf16*)(ws + WS_HCNBF);
    float*  rownorm = (float*)(ws + WS_ROWNORM);
    __bf16* fc1bf   = (__bf16*)(ws + WS_FC1BF);
    __bf16* fc2bf   = (__bf16*)(ws + WS_FC2BF);
    __bf16* coswn   = (__bf16*)(ws + WS_COSWN);

    // zero the reduction stats region
    hipMemsetAsync(stats, 0, 3584 * sizeof(float), stream);

    // weight prep
    f2bf_kernel<<<256, 256, 0, stream>>>(fc1_w, fc1bf, 65536);
    f2bf_kernel<<<256, 256, 0, stream>>>(fc2_w, fc2bf, 65536);
    cosw_norm_kernel<<<256, 256, 0, stream>>>(cos_W, coswn);

    const int nblk = (N_NODES + 255) / 256;   // 391

    // conv stage
    conv1_kernel<<<nblk, 256, 0, stream>>>(x, w1, b1, y1, stats);
    finalize_bn_kernel<<<1, 64, 0, stream>>>(stats + ST_BN1, bn1_g, bn1_b, 32,
                                             (float)N_NODES * 12.f);
    conv2_kernel<<<N_NODES / 4, 256, 0, stream>>>(y1, w2, b2, stats, featbuf, stats);
    finalize_bn_kernel<<<1, 64, 0, stream>>>(stats + ST_BN2, bn2_g, bn2_b, 64,
                                             (float)N_NODES * 4.f);
    bn2relu_bf_kernel<<<(int)(((size_t)N_NODES * 256 + 255) / 256), 256, 0, stream>>>(
        featbuf, stats, inbf);

    dim3 gemm_grid(256 / 256, N_NODES / 32);   // (1, 3125)

    // --- GAT layer 1 ---
    wmma_gemm_kernel<<<gemm_grid, 256, 0, stream>>>(inbf, fc1bf, featbuf, 256, 256, nullptr, 1.0f);
    eler_kernel<<<nblk, 256, 0, stream>>>(featbuf, attn1l, attn1r, elerbuf);
    gat_aggregate_kernel<<<N_NODES, 128, 0, stream>>>(featbuf, elerbuf, src, bias1,
                                                      hc + 0, inbf, 1);
    // --- GAT layer 2 ---
    wmma_gemm_kernel<<<gemm_grid, 256, 0, stream>>>(inbf, fc2bf, featbuf, 256, 256, nullptr, 1.0f);
    eler_kernel<<<nblk, 256, 0, stream>>>(featbuf, attn2l, attn2r, elerbuf);
    gat_aggregate_kernel<<<N_NODES, 128, 0, stream>>>(featbuf, elerbuf, src, bias2,
                                                      hc + 256, inbf, 2);
    // --- GAT layer 3 (reuses fc2/attn2/bias2) ---
    wmma_gemm_kernel<<<gemm_grid, 256, 0, stream>>>(inbf, fc2bf, featbuf, 256, 256, nullptr, 1.0f);
    eler_kernel<<<nblk, 256, 0, stream>>>(featbuf, attn2l, attn2r, elerbuf);
    gat_aggregate_kernel<<<N_NODES, 128, 0, stream>>>(featbuf, elerbuf, src, bias2,
                                                      hc + 512, nullptr, 3);

    // concat batch-norm over nodes
    dim3 cs_grid(3, (N_NODES + 255) / 256);
    colstats_kernel<<<cs_grid, 256, 0, stream>>>(hc, stats);
    finalize_bn_kernel<<<1, 768, 0, stream>>>(stats + ST_DBN, dbn_g, dbn_b, 768, (float)N_NODES);
    hcn_rownorm_kernel<<<N_NODES, 256, 0, stream>>>(hc, stats, hcnbf, rownorm);

    // cosine head: (N x 768) @ coswn^T, per-row scale COS_S / ||row||
    wmma_gemm_kernel<<<gemm_grid, 256, 0, stream>>>(hcnbf, coswn, (float*)d_out, 768, 256,
                                                    rownorm, COS_S);
}
